// O3AttentionLayer_81381040324943
// MI455X (gfx1250) — compile-verified, matrix operand order checked by reading, not run on hardware
//
#include <hip/hip_runtime.h>
#include <hip/hip_bf16.h>
#include <stdint.h>

typedef __attribute__((ext_vector_type(16))) __bf16         v16bf;
typedef __attribute__((ext_vector_type(16))) unsigned short v16us;
typedef __attribute__((ext_vector_type(8)))  float          v8f;
typedef __attribute__((ext_vector_type(8)))  unsigned int   v8u;
typedef __attribute__((ext_vector_type(4)))  unsigned int   u32x4;
typedef __attribute__((ext_vector_type(8)))  int            i32x8;
typedef __attribute__((ext_vector_type(4)))  int            i32x4;

#define RMAXF      2.5f
#define SILU_C     1.6765849f        /* 1/sqrt(E[silu(z)^2]), z~N(0,1) */
#define INV3       0.57735026919f
#define INV6       0.40824829046f
#define PW0        0.20412414523f    /* sqrt(1/24)  */
#define PW1        0.30618621785f    /* sqrt(3/32)  */
#define PWSIM      0.05590169944f    /* sqrt(1/320) */
#define RBSCALE    5.0596442563f     /* sqrt(2/2.5)*sqrt(32) */
#define THETA_K    1.2566370614f     /* pi/2.5 */
#define INV_SQRT32 0.17677669530f
#define INV_SQRT8  0.35355339059f

// ---------- helpers ----------
__device__ __forceinline__ unsigned short f2bf(float f) {
  union { __bf16 b; unsigned short u; } cv;
  cv.b = (__bf16)f;                       // native f32->bf16 convert on gfx1250
  return cv.u;
}
__device__ __forceinline__ uint32_t pack_bf16(float a, float b) {
  return ((uint32_t)f2bf(b) << 16) | (uint32_t)f2bf(a);
}
// monotone float <-> uint for atomicMax-based segment max
__device__ __forceinline__ unsigned int fenc(float x) {
  unsigned int u = __float_as_uint(x);
  return (u & 0x80000000u) ? ~u : (u | 0x80000000u);
}
__device__ __forceinline__ float fdec(unsigned int e) {
  unsigned int u = (e & 0x80000000u) ? (e & 0x7FFFFFFFu) : ~e;
  return __uint_as_float(u);
}

// ---------------------------------------------------------------------------
// Kernel 1: per-node pre-contracted queries  qs[n] = [q0@w_sim0 (16) | (q1 w_sim1) (8x3)]
// ---------------------------------------------------------------------------
__global__ void node_q_kernel(const float* __restrict__ x,
                              const float* __restrict__ wq0,
                              const float* __restrict__ wq1,
                              const float* __restrict__ ws0,
                              const float* __restrict__ ws1,
                              float* __restrict__ qs, int N) {
  __shared__ float sq0[256], ss0[256], sq1[64], ss1[64];
  for (int i = threadIdx.x; i < 256; i += blockDim.x) { sq0[i] = wq0[i] * 0.25f; ss0[i] = ws0[i]; }
  for (int i = threadIdx.x; i < 64;  i += blockDim.x) { sq1[i] = wq1[i] * INV_SQRT8; ss1[i] = ws1[i]; }
  __syncthreads();
  int n = blockIdx.x * blockDim.x + threadIdx.x;
  if (n >= N) return;
  const float* row = x + (size_t)n * 40;
  float x0[16], x1[8][3];
#pragma unroll
  for (int u = 0; u < 16; ++u) x0[u] = row[u];
#pragma unroll
  for (int u = 0; u < 8; ++u) {
    x1[u][0] = row[16 + u*3 + 0]; x1[u][1] = row[16 + u*3 + 1]; x1[u][2] = row[16 + u*3 + 2];
  }
  float q0[16];
#pragma unroll
  for (int v = 0; v < 16; ++v) {
    float a = 0.f;
#pragma unroll
    for (int u = 0; u < 16; ++u) a += x0[u] * sq0[u*16 + v];
    q0[v] = a;
  }
  float q1[8][3];
#pragma unroll
  for (int v = 0; v < 8; ++v) {
#pragma unroll
    for (int i = 0; i < 3; ++i) {
      float a = 0.f;
#pragma unroll
      for (int u = 0; u < 8; ++u) a += x1[u][i] * sq1[u*8 + v];
      q1[v][i] = a;
    }
  }
  float* o = qs + (size_t)n * 40;
#pragma unroll
  for (int v = 0; v < 16; ++v) {
    float a = 0.f;
#pragma unroll
    for (int u = 0; u < 16; ++u) a += q0[u] * ss0[u*16 + v];
    o[v] = a;
  }
#pragma unroll
  for (int v = 0; v < 8; ++v) {
#pragma unroll
    for (int i = 0; i < 3; ++i) {
      float a = 0.f;
#pragma unroll
      for (int u = 0; u < 8; ++u) a += q1[u][i] * ss1[u*8 + v];
      o[16 + v*3 + i] = a;
    }
  }
}

// ---------------------------------------------------------------------------
// Kernel 2: heavy per-edge kernel. 1 wave == 16 edges.
//   WMMA chain: rb(16x32 bf16) x W1(32x16) -> silu -> h (pad K to 32)
//               h x W2-tile(16x16) x 40 tiles, software-pipelined through a
//               double-buffered LDS transpose, streamed into fc_tp accumulators.
// ---------------------------------------------------------------------------
__launch_bounds__(128, 1)
__global__ void edge_main_kernel(const float* __restrict__ x,
                                 const float* __restrict__ pos,
                                 const int*   __restrict__ esrc,
                                 const int*   __restrict__ edst,
                                 const float* __restrict__ wk1,
                                 const float* __restrict__ wk2,
                                 const float* __restrict__ wv1,
                                 const float* __restrict__ wv2,
                                 const float* __restrict__ qs,
                                 float* __restrict__ sarr,
                                 float* __restrict__ vals,
                                 unsigned int* __restrict__ smax,
                                 int E, int G) {
  // [path][tile][col lane][K-pair dword], bf16 packed, scaled by 1/sqrt(16)
  __shared__ __align__(16) uint32_t w2t[2][40][16][8];          // 40 KB
  // [path][lane][K-pair dword] B-layout of W1/sqrt(32)
  __shared__ __align__(16) uint32_t w1b[2][32][8];              // 2 KB
  // raw f32 W1 staged by the Tensor Data Mover
  __shared__ __align__(16) float    w1raw[2][512];              // 4 KB
  // per-wave double-buffered transpose scratch, stride 20 floats
  __shared__ __align__(16) float    scratch[4][2][16][20];      // 10 KB

  // ---- stage W1 into LDS with the TDM (async tensor DMA) ----
#if __has_builtin(__builtin_amdgcn_tensor_load_to_lds)
  if (threadIdx.x < 32) {                       // one wave issues the DMAs
#pragma unroll
    for (int p = 0; p < 2; ++p) {
      const float* src = p ? wv1 : wk1;
      unsigned long long ga = (unsigned long long)(uintptr_t)src;
      unsigned int lds_off  = (unsigned int)(uintptr_t)&w1raw[p][0];
      const int dim0 = 512;                     // 32x16 f32 as one row
      u32x4 g0;
      g0[0] = 1u;                                              // count=1 (valid user D#)
      g0[1] = lds_off;                                         // lds_addr (bytes)
      g0[2] = (unsigned int)(ga & 0xFFFFFFFFu);                // global_addr[31:0]
      g0[3] = (unsigned int)((ga >> 32) & 0x01FFFFFFu) | (2u << 30); // addr[56:32], type=2
      i32x8 g1;
      g1[0] = (2 << 16);                                       // data_size = 4B
      g1[1] = (dim0 & 0xFFFF) << 16;                           // tensor_dim0[15:0]
      g1[2] = ((dim0 >> 16) & 0xFFFF) | (1 << 16);             // tensor_dim0[31:16], tensor_dim1 lo=1
      g1[3] = (dim0 & 0xFFFF) << 16;                           // tensor_dim1 hi=0, tile_dim0=512
      g1[4] = 1;                                               // tile_dim1=1, tile_dim2=0
      g1[5] = dim0;                                            // tensor_dim0_stride[31:0]
      g1[6] = 0;
      g1[7] = 0;
      i32x4 g2 = {0, 0, 0, 0}, g3 = {0, 0, 0, 0};              // dims 2..4 unused
#if __clang_major__ >= 23
      i32x8 g23 = {0, 0, 0, 0, 0, 0, 0, 0};
      __builtin_amdgcn_tensor_load_to_lds(g0, g1, g2, g3, g23, 0);
#else
      __builtin_amdgcn_tensor_load_to_lds(g0, g1, g2, g3, 0);
#endif
    }
    __builtin_amdgcn_s_wait_tensorcnt(0);
  }
#else
  for (int i = threadIdx.x; i < 1024; i += blockDim.x)
    w1raw[i >> 9][i & 511] = (i >> 9) ? wv1[i & 511] : wk1[i & 511];
#endif
  __syncthreads();

  // ---- cooperative pack of shared weights (W1 from LDS, W2 from L2) ----
  for (int idx = threadIdx.x; idx < 2*40*16*8; idx += blockDim.x) {
    int p   = idx / 5120;
    int rem = idx - p * 5120;
    int nt  = rem >> 7;
    int n   = (rem >> 3) & 15;
    int j   = rem & 7;
    const float* W2 = p ? wv2 : wk2;
    int c = nt*16 + n;
    w2t[p][nt][n][j] = pack_bf16(W2[(2*j)*640 + c] * 0.25f, W2[(2*j+1)*640 + c] * 0.25f);
  }
  for (int idx = threadIdx.x; idx < 2*32*8; idx += blockDim.x) {
    int p = idx >> 8;
    int l = (idx >> 3) & 31;
    int j = idx & 7;
    int n = l & 15, kb = (l >> 4) * 16;
    w1b[p][l][j] = pack_bf16(w1raw[p][(kb + 2*j)*16 + n] * INV_SQRT32,
                             w1raw[p][(kb + 2*j + 1)*16 + n] * INV_SQRT32);
  }
  __syncthreads();

  const int wv   = threadIdx.x >> 5;
  const int lane = threadIdx.x & 31;
  const int g    = blockIdx.x * 4 + wv;
  if (g >= G) return;
  const int el   = lane & 15;     // edge slot within the group
  const int half = lane >> 4;     // 0: K 0..15 side, 1: K 16..31 side
  const int e    = g * 16 + el;
  const bool live = (e < E);
  const int es = live ? esrc[e] : 0;
  const int ed = live ? edst[e] : 0;

  // ---- geometry + radial basis (sine recurrence) ----
  float vx = pos[es*3+0] - pos[ed*3+0];
  float vy = pos[es*3+1] - pos[ed*3+1];
  float vz = pos[es*3+2] - pos[ed*3+2];
  float r  = sqrtf(vx*vx + vy*vy + vz*vz + 1e-12f);
  float rinv = __builtin_amdgcn_rcpf(r);
  float sh1x = 1.7320508f * vx * rinv, sh1y = 1.7320508f * vy * rinv, sh1z = 1.7320508f * vz * rinv;
  float scale = (live && r < RMAXF) ? (RBSCALE * rinv) : 0.f;
  float th = THETA_K * r, s1, c1;
  __sincosf(th, &s1, &c1);
  float c2 = 2.f * c1;
  float lo[16], hi[16];
  {
    float sp = 0.f, sc = s1;
#pragma unroll
    for (int n = 1; n <= 32; ++n) {
      float v = sc * scale;
      if (n <= 16) lo[n-1] = v; else hi[n-17] = v;
      float nx = c2 * sc - sp; sp = sc; sc = nx;
    }
  }
  v16bf arb;
  {
    v16us tmp;
#pragma unroll
    for (int j = 0; j < 16; ++j) tmp[j] = f2bf(half ? hi[j] : lo[j]);
    arb = __builtin_bit_cast(v16bf, tmp);
  }

  // ---- gather source features (L2-resident) ----
  const float* xr = x + (size_t)es * 40;
  __builtin_prefetch(xr, 0, 3);
  __builtin_prefetch(qs + (size_t)ed * 40, 0, 3);
  float x0s[16], x1s[8][3], dotv[8], crx[8][3];
#pragma unroll
  for (int u = 0; u < 16; ++u) x0s[u] = xr[u];
#pragma unroll
  for (int u = 0; u < 8; ++u) {
    float a = xr[16 + u*3 + 0], b = xr[16 + u*3 + 1], c = xr[16 + u*3 + 2];
    x1s[u][0] = a; x1s[u][1] = b; x1s[u][2] = c;
    dotv[u]   = a*sh1x + b*sh1y + c*sh1z;
    crx[u][0] = b*sh1z - c*sh1y;
    crx[u][1] = c*sh1x - a*sh1z;
    crx[u][2] = a*sh1y - b*sh1x;
  }

  float* buf0 = &scratch[wv][0][0][0];   // stride-20 rows
  float* buf1 = &scratch[wv][1][0][0];

#pragma unroll
  for (int p = 0; p < 2; ++p) {      // 0 = key path, 1 = value path
    // -------- layer 1 WMMA: rb x W1 --------
    v16bf b1 = __builtin_bit_cast(v16bf, *(const v8u*)&w1b[p][lane][0]);
    v8f d1 = {};
    d1 = __builtin_amdgcn_wmma_f32_16x16x32_bf16(false, arb, false, b1, (short)0, d1, false, false);

    // silu (fast rcp), then C-layout -> A-layout transpose through LDS
#pragma unroll
    for (int gi = 0; gi < 8; ++gi) {
      float hv = d1[gi];
      hv = SILU_C * hv * __builtin_amdgcn_rcpf(1.f + __expf(-hv));
      buf0[(gi + 8*half)*20 + el] = hv;
    }
    v16bf ah;
    {
      v16us tmp;
      if (half == 0) {
        const float* hrow = buf0 + el*20;   // in-order LDS: RAW through LDS is safe
#pragma unroll
        for (int j = 0; j < 16; ++j) tmp[j] = f2bf(hrow[j]);
      } else {
#pragma unroll
        for (int j = 0; j < 16; ++j) tmp[j] = 0;   // zero pad K = 16..31
      }
      ah = __builtin_bit_cast(v16bf, tmp);
    }

    // -------- layer 2: 40 WMMA tiles, software-pipelined by one tile --------
    float o0[16], o1[8][3];
#pragma unroll
    for (int i = 0; i < 16; ++i) o0[i] = 0.f;
#pragma unroll
    for (int w = 0; w < 8; ++w) { o1[w][0] = 0.f; o1[w][1] = 0.f; o1[w][2] = 0.f; }

    // prologue: tile 0
    {
      v16bf bt;
      if (half == 0) bt = __builtin_bit_cast(v16bf, *(const v8u*)&w2t[p][0][el][0]);
      else { v8u z = {}; bt = __builtin_bit_cast(v16bf, z); }
      v8f dt = {};
      dt = __builtin_amdgcn_wmma_f32_16x16x32_bf16(false, ah, false, bt, (short)0, dt, false, false);
#pragma unroll
      for (int gi = 0; gi < 8; ++gi) buf0[(gi + 8*half)*20 + el] = dt[gi];
    }

#pragma unroll
    for (int nt = 1; nt <= 40; ++nt) {
      const int ct = nt - 1;                      // tile being consumed
      float* cbuf = (ct & 1) ? buf1 : buf0;
      float* nbuf = (nt & 1) ? buf1 : buf0;

      // issue loads of tile ct first (their latency hides behind the next WMMA)
      float t[16];
      {
        const float4* rp = (const float4*)(cbuf + el*20);
        float4 a0 = rp[0], a1 = rp[1], a2 = rp[2], a3 = rp[3];
        t[0]=a0.x; t[1]=a0.y; t[2]=a0.z; t[3]=a0.w;
        t[4]=a1.x; t[5]=a1.y; t[6]=a1.z; t[7]=a1.w;
        t[8]=a2.x; t[9]=a2.y; t[10]=a2.z; t[11]=a2.w;
        t[12]=a3.x; t[13]=a3.y; t[14]=a3.z; t[15]=a3.w;
      }

      if (nt < 40) {
        v16bf bt;
        if (half == 0) bt = __builtin_bit_cast(v16bf, *(const v8u*)&w2t[p][nt][el][0]);
        else { v8u z = {}; bt = __builtin_bit_cast(v16bf, z); }
        v8f dt = {};
        dt = __builtin_amdgcn_wmma_f32_16x16x32_bf16(false, ah, false, bt, (short)0, dt, false, false);
#pragma unroll
        for (int gi = 0; gi < 8; ++gi) nbuf[(gi + 8*half)*20 + el] = dt[gi];
      }

      // compile-time decode of columns c = 16*ct + n into (matrix, u, w)
      if (ct < 16) {                       // w0: out0[w] += x0[u]*W
#pragma unroll
        for (int n = 0; n < 16; ++n) o0[n] += x0s[ct] * t[n];
      } else if (ct < 24) {                // w1: out1[w][i] += INV3*x0[u]*sh1[i]*W
        const int u0 = 2*(ct - 16);
#pragma unroll
        for (int n = 0; n < 16; ++n) {
          const int u = u0 + (n >> 3), w = n & 7;
          float tmp = INV3 * x0s[u] * t[n];
          o1[w][0] += tmp * sh1x; o1[w][1] += tmp * sh1y; o1[w][2] += tmp * sh1z;
        }
      } else if (ct < 28) {                // w2: out1[w][i] += INV3*x1[u][i]*W
        const int u0 = 2*(ct - 24);
#pragma unroll
        for (int n = 0; n < 16; ++n) {
          const int u = u0 + (n >> 3), w = n & 7;
          float tv = INV3 * t[n];
          o1[w][0] += tv * x1s[u][0]; o1[w][1] += tv * x1s[u][1]; o1[w][2] += tv * x1s[u][2];
        }
      } else if (ct < 36) {                // w3: out0[w] += INV3*dot[u]*W
#pragma unroll
        for (int n = 0; n < 16; ++n) o0[n] += INV3 * dotv[ct - 28] * t[n];
      } else {                             // w4: out1[w][i] += INV6*crx[u][i]*W
        const int u0 = 2*(ct - 36);
#pragma unroll
        for (int n = 0; n < 16; ++n) {
          const int u = u0 + (n >> 3), w = n & 7;
          float tv = INV6 * t[n];
          o1[w][0] += tv * crx[u][0]; o1[w][1] += tv * crx[u][1]; o1[w][2] += tv * crx[u][2];
        }
      }
    } // nt

    if (p == 0) {
      // ---- key path: attention logit against pre-contracted query ----
      if (half == 0 && live) {
        const float* qr = qs + (size_t)ed * 40;
        float s0 = 0.f;
#pragma unroll
        for (int v = 0; v < 16; ++v) s0 += qr[v] * (PW0 * o0[v]);
        float s1a = 0.f;
#pragma unroll
        for (int w = 0; w < 8; ++w) {
          s1a += qr[16 + w*3 + 0] * (PW1 * o1[w][0]);
          s1a += qr[16 + w*3 + 1] * (PW1 * o1[w][1]);
          s1a += qr[16 + w*3 + 2] * (PW1 * o1[w][2]);
        }
        float s = PWSIM * (s0 + INV3 * s1a);
        sarr[e] = s;
        atomicMax(smax + ed, fenc(s));
      }
    } else {
      // ---- value path: stage 16x40 in LDS (flat), store coalesced ----
      float* stage = buf0;                 // 640 floats available (both buffers)
      if (half == 0) {
#pragma unroll
        for (int v = 0; v < 16; ++v) stage[el*40 + v] = PW0 * o0[v];
#pragma unroll
        for (int w = 0; w < 8; ++w) {
          stage[el*40 + 16 + w*3 + 0] = PW1 * o1[w][0];
          stage[el*40 + 16 + w*3 + 1] = PW1 * o1[w][1];
          stage[el*40 + 16 + w*3 + 2] = PW1 * o1[w][2];
        }
      }
      const int e0 = g * 16;
#pragma unroll
      for (int j = 0; j < 20; ++j) {
        int idx = lane + 32*j;             // 0..639
        int ee = idx / 40, cc = idx - ee*40;
        if (e0 + ee < E) vals[(size_t)(e0 + ee)*40 + cc] = stage[ee*40 + cc];
      }
    }
  } // path loop
}

// ---------------------------------------------------------------------------
// Kernel 3: exp(s - smax[dst]) and segment sums
// ---------------------------------------------------------------------------
__global__ void edge_ex_kernel(const int* __restrict__ edst,
                               const unsigned int* __restrict__ smax,
                               float* __restrict__ sarr,
                               float* __restrict__ ssum, int E) {
  int e = blockIdx.x * blockDim.x + threadIdx.x;
  if (e >= E) return;
  int d = edst[e];
  float ex = __expf(sarr[e] - fdec(smax[d]));
  sarr[e] = ex;
  atomicAdd(ssum + d, ex);
}

// ---------------------------------------------------------------------------
// Kernel 4: coef = sqrt(attn); scatter-add values into out[dst]
//           8 threads per edge -> coalesced column atomics
// ---------------------------------------------------------------------------
__global__ void edge_scatter_kernel(const int* __restrict__ edst,
                                    const float* __restrict__ sarr,
                                    const float* __restrict__ ssum,
                                    const float* __restrict__ vals,
                                    float* __restrict__ out, int E) {
  int idx = blockIdx.x * blockDim.x + threadIdx.x;
  int e = idx >> 3, c8 = idx & 7;
  if (e >= E) return;
  int d = edst[e];
  float coef = sqrtf(sarr[e] * __builtin_amdgcn_rcpf(ssum[d]));
  const float* vr = vals + (size_t)e * 40;
  float* orow = out + (size_t)d * 40;
#pragma unroll
  for (int k = 0; k < 5; ++k) {
    int c = c8 + 8*k;
    atomicAdd(orow + c, coef * vr[c]);
  }
}

// ---------------------------------------------------------------------------
extern "C" void kernel_launch(void* const* d_in, const int* in_sizes, int n_in,
                              void* d_out, int out_size, void* d_ws, size_t ws_size,
                              hipStream_t stream) {
  const float* x    = (const float*)d_in[0];
  const float* pos  = (const float*)d_in[1];
  const int*   esrc = (const int*)d_in[2];
  const int*   edst = (const int*)d_in[3];
  const float* wk1  = (const float*)d_in[4];
  const float* wk2  = (const float*)d_in[5];
  const float* wv1  = (const float*)d_in[6];
  const float* wv2  = (const float*)d_in[7];
  const float* wq0  = (const float*)d_in[8];
  const float* wq1  = (const float*)d_in[9];
  const float* ws0  = (const float*)d_in[10];
  const float* ws1  = (const float*)d_in[11];
  const int N = in_sizes[0] / 40;
  const int E = in_sizes[2];

  // workspace layout (floats)
  float* qs   = (float*)d_ws;                         // N*40
  float* sarr = qs + (size_t)N * 40;                  // E
  float* vals = sarr + (size_t)E;                     // E*40
  float* ssum = vals + (size_t)E * 40;                // N
  unsigned int* smax = (unsigned int*)(ssum + N);     // N

  hipMemsetAsync(d_out, 0, (size_t)out_size * sizeof(float), stream);
  hipMemsetAsync(ssum, 0, (size_t)N * sizeof(float), stream);
  hipMemsetAsync(smax, 0, (size_t)N * sizeof(unsigned int), stream);

  node_q_kernel<<<(N + 127) / 128, 128, 0, stream>>>(x, wq0, wq1, ws0, ws1, qs, N);

  const int G = (E + 15) / 16;      // 16-edge groups, one per wave
  edge_main_kernel<<<(G + 3) / 4, 128, 0, stream>>>(x, pos, esrc, edst,
                                                    wk1, wk2, wv1, wv2,
                                                    qs, sarr, vals, smax, E, G);

  edge_ex_kernel<<<(E + 255) / 256, 256, 0, stream>>>(edst, smax, sarr, ssum, E);
  edge_scatter_kernel<<<((E * 8) + 255) / 256, 256, 0, stream>>>(edst, sarr, ssum, vals,
                                                                 (float*)d_out, E);
}
